// NeRF_MLP_55997783605497
// MI455X (gfx1250) — matrix operand order repaired and checked by reading, not compiled, not run on hardware
//
#include <hip/hip_runtime.h>
#include <hip/hip_bf16.h>

typedef __attribute__((ext_vector_type(16))) _Float16 v16h;
typedef __attribute__((ext_vector_type(8)))  _Float16 v8h;
typedef __attribute__((ext_vector_type(2)))  _Float16 v2h;
typedef __attribute__((ext_vector_type(2)))  __fp16   v2fp16;
typedef __attribute__((ext_vector_type(8)))  float    v8f;

#define ROWS 64   // rows per block (4 M-tiles of 16)

// ---------------------------------------------------------------------------
// Pack fp32 W[Nout][Kreal] -> fp16 fragments in WMMA B-operand lane order.
// Fragment (ntile, kfrag): 32 lanes x 16 halves (32B) contiguous per lane.
// Lane l supplies column n = ntile*16 + (l&15); dword j holds
// K = kfrag*32 + ((l>>4)<<4) + 2j .. +1, zero-padded past Kreal.
// ---------------------------------------------------------------------------
__global__ __launch_bounds__(256) void pack_weights(
    const float* __restrict__ W, unsigned* __restrict__ dst,
    int Kreal, int Kpad, int Nout)
{
    const int nk = Kpad >> 5;
    const int total = (Nout >> 4) * nk * 256;      // dwords
    int idx = blockIdx.x * 256 + threadIdx.x;
    if (idx >= total) return;
    const int j     = idx & 7;
    const int lane  = (idx >> 3) & 31;
    const int frag  = idx >> 8;
    const int ntile = frag / nk;
    const int k0    = (frag % nk) << 5;
    const int n = ntile * 16 + (lane & 15);
    const int k = k0 + ((lane >> 4) << 4) + 2 * j;
    const float f0 = (k     < Kreal) ? W[(size_t)n * Kreal + k]     : 0.0f;
    const float f1 = (k + 1 < Kreal) ? W[(size_t)n * Kreal + k + 1] : 0.0f;
    const unsigned short h0 = __builtin_bit_cast(unsigned short, (_Float16)f0);
    const unsigned short h1 = __builtin_bit_cast(unsigned short, (_Float16)f1);
    dst[idx] = (unsigned)h0 | ((unsigned)h1 << 16);
}

// ---------------------------------------------------------------------------
// Each wave owns NT disjoint N-tiles and ALL 4 M-tiles (8 live accumulators,
// zero-initialized so the first WMMA folds C to inline 0).
// kf outer loop: 4 A frags (LDS, reused NT times) + NT B frags (global,
// reused 4 times) feed 4*NT independent back-to-back WMMAs.
// Across the 8 waves of the block, B fragments are loaded exactly once.
// Bias is added in the epilogue (f32), then pk-convert + pk-relu.
// ---------------------------------------------------------------------------
template<int NT>
__device__ __forceinline__ void layer_wave(
    int wave, int lane,
    const _Float16* __restrict__ Ain, int astride,
    _Float16* __restrict__ Aout, int ostride,
    const _Float16* __restrict__ Wp, const float* __restrict__ bias,
    int nk, bool relu)
{
    const int m     = lane & 15;
    const int hi    = lane >> 4;
    const int koffA = hi << 4;            // A lane K offset (0 or 16)
    const int ncol  = lane & 15;

    v8f acc[4][NT];
    {
        v8f z = {};
        #pragma unroll
        for (int t = 0; t < NT; ++t)
            #pragma unroll
            for (int mt = 0; mt < 4; ++mt) acc[mt][t] = z;
    }

    // This wave's fragment stream: tile t at +t*nk*512 halves, kf at +kf*512.
    const _Float16* wbase = Wp + ((size_t)(wave * NT) * nk * 32 + lane) * 16;

    for (int kf = 0; kf < nk; ++kf) {
        union AU { v16h v; v8h h[2]; };

        AU a[4];
        #pragma unroll
        for (int mt = 0; mt < 4; ++mt) {
            const _Float16* arow = Ain + (mt * 16 + m) * astride + (kf << 5) + koffA;
            a[mt].h[0] = *(const v8h*)(arow);        // ds_load_b128
            a[mt].h[1] = *(const v8h*)(arow + 8);    // ds_load_b128
        }

        AU b[NT];
        #pragma unroll
        for (int t = 0; t < NT; ++t)
            b[t].v = *(const v16h*)(wbase + (size_t)t * nk * 512 + (size_t)kf * 512);

        #pragma unroll
        for (int t = 0; t < NT; ++t)
            #pragma unroll
            for (int mt = 0; mt < 4; ++mt)
                acc[mt][t] = __builtin_amdgcn_wmma_f32_16x16x32_f16(
                                false, a[mt].v, false, b[t].v, (short)0,
                                acc[mt][t], false, false);
    }

    #pragma unroll
    for (int t = 0; t < NT; ++t) {
        const int nbase = (wave * NT + t) << 4;
        const float bv = bias[nbase + ncol];
        #pragma unroll
        for (int mt = 0; mt < 4; ++mt) {
            #pragma unroll
            for (int r2 = 0; r2 < 4; ++r2) {
                const float v0 = acc[mt][t][2 * r2]     + bv;
                const float v1 = acc[mt][t][2 * r2 + 1] + bv;
                v2h p = __builtin_bit_cast(v2h,
                            __builtin_amdgcn_cvt_pkrtz(v0, v1)); // v_cvt_pk_f16_f32
                if (relu) {
                    const v2h z = { (_Float16)0.0f, (_Float16)0.0f };
                    p = __builtin_elementwise_max(p, z);         // v_pk_max_num_f16
                }
                const int mm = mt * 16 + (hi << 3) + 2 * r2;     // D rows mm, mm+1
                Aout[ mm      * ostride + nbase + ncol] = p.x;
                Aout[(mm + 1) * ostride + nbase + ncol] = p.y;
            }
        }
    }
}

__global__ __launch_bounds__(256) void nerf_mlp_fused(
    const float* __restrict__ xyz, const float* __restrict__ dvec,
    const _Float16* __restrict__ P1, const _Float16* __restrict__ P2,
    const _Float16* __restrict__ P3, const _Float16* __restrict__ P4,
    const _Float16* __restrict__ P5, const _Float16* __restrict__ P6,
    const _Float16* __restrict__ P7, const _Float16* __restrict__ P8,
    const _Float16* __restrict__ P9, const _Float16* __restrict__ P10,
    const float* __restrict__ b1,  const float* __restrict__ b2,
    const float* __restrict__ b3,  const float* __restrict__ b4,
    const float* __restrict__ b5,  const float* __restrict__ b6,
    const float* __restrict__ b7,  const float* __restrict__ b8,
    const float* __restrict__ b9,  const float* __restrict__ b10,
    const float* __restrict__ W9,  const float* __restrict__ W11,
    const float* __restrict__ b11,
    float* __restrict__ outColor, float* __restrict__ outDensity)
{
    __shared__ __align__(16) _Float16 bufA[ROWS][320];
    __shared__ __align__(16) _Float16 bufB[ROWS][320];
    __shared__ __align__(16) _Float16 xemb[ROWS][64];   // 63 real + 1 zero pad
    __shared__ __align__(16) _Float16 demb[ROWS][32];   // 27 real + 5 zero pad

    const int tid  = threadIdx.x;
    const int wave = tid >> 5;
    const int lane = tid & 31;
    const int rowbase = blockIdx.x * ROWS;

    // ---- positional encodings (VALU trans ops) ----
    for (int idx = tid; idx < ROWS * 64; idx += 256) {
        const int mrow = idx >> 6, j = idx & 63;
        float v = 0.0f;
        if (j < 3) {
            v = xyz[(rowbase + mrow) * 3 + j];
        } else if (j < 63) {
            const int jj = j - 3;
            const int l = jj / 6, rem = jj % 6;
            const int s = rem / 3, dim = rem % 3;
            const float t = xyz[(rowbase + mrow) * 3 + dim] * (float)(1 << l);
            v = s ? __cosf(t) : __sinf(t);
        }
        xemb[mrow][j] = (_Float16)v;
    }
    for (int idx = tid; idx < ROWS * 32; idx += 256) {
        const int mrow = idx >> 5, j = idx & 31;
        float v = 0.0f;
        if (j < 3) {
            v = dvec[(rowbase + mrow) * 3 + j];
        } else if (j < 27) {
            const int jj = j - 3;
            const int l = jj / 6, rem = jj % 6;
            const int s = rem / 3, dim = rem % 3;
            const float t = dvec[(rowbase + mrow) * 3 + dim] * (float)(1 << l);
            v = s ? __cosf(t) : __sinf(t);
        }
        demb[mrow][j] = (_Float16)v;
    }
    __syncthreads();

    // ---- trunk MLP on WMMA ----
    layer_wave<2>(wave, lane, &xemb[0][0],  64, &bufA[0][0], 320, P1, b1,  2, true);
    __syncthreads();
    layer_wave<2>(wave, lane, &bufA[0][0], 320, &bufB[0][0], 320, P2, b2,  8, true);
    __syncthreads();
    layer_wave<2>(wave, lane, &bufB[0][0], 320, &bufA[0][0], 320, P3, b3,  8, true);
    __syncthreads();
    layer_wave<2>(wave, lane, &bufA[0][0], 320, &bufB[0][0], 320, P4, b4,  8, true);
    __syncthreads();
    layer_wave<2>(wave, lane, &bufB[0][0], 320, &bufA[0][0], 320, P5, b5,  8, true);
    __syncthreads();

    // skip concat: [act5 | x_emb] -> bufA cols 256..319 (col 319 = pad = 0)
    for (int idx = tid; idx < ROWS * 64; idx += 256) {
        const int mrow = idx >> 6, j = idx & 63;
        bufA[mrow][256 + j] = xemb[mrow][j];
    }
    __syncthreads();

    layer_wave<2>(wave, lane, &bufA[0][0], 320, &bufB[0][0], 320, P6, b6, 10, true);
    __syncthreads();
    layer_wave<2>(wave, lane, &bufB[0][0], 320, &bufA[0][0], 320, P7, b7,  8, true);
    __syncthreads();
    layer_wave<2>(wave, lane, &bufA[0][0], 320, &bufB[0][0], 320, P8, b8,  8, true);
    __syncthreads();

    // layer 9: first 256 outputs, NO activation
    layer_wave<2>(wave, lane, &bufB[0][0], 320, &bufA[0][0], 320, P9, b9,  8, false);
    __syncthreads();

    // density head: relu(act8 . W9[256] + b9[256])  (bufB still holds act8)
    for (int r = tid; r < ROWS; r += 256) {
        const float* w = W9 + 256 * 256;
        float acc = b9[256];
        for (int k = 0; k < 256; ++k) acc += (float)bufB[r][k] * w[k];
        outDensity[rowbase + r] = fmaxf(acc, 0.0f);
    }
    // view concat: [x9 | d_emb] -> bufA cols 256..287 (283..287 = 0 pad)
    for (int idx = tid; idx < ROWS * 32; idx += 256) {
        const int mrow = idx >> 5, j = idx & 31;
        bufA[mrow][256 + j] = demb[mrow][j];
    }
    __syncthreads();

    layer_wave<1>(wave, lane, &bufA[0][0], 320, &bufB[0][0], 320, P10, b10, 9, true);
    __syncthreads();

    // color head: sigmoid(W11 @ act10 + b11)
    for (int idx = tid; idx < ROWS * 3; idx += 256) {
        const int mrow = idx / 3, o = idx % 3;
        const float* w = W11 + o * 128;
        float acc = b11[o];
        for (int k = 0; k < 128; ++k) acc += (float)bufB[mrow][k] * w[k];
        outColor[(rowbase + mrow) * 3 + o] = 1.0f / (1.0f + __expf(-acc));
    }
}

extern "C" void kernel_launch(void* const* d_in, const int* in_sizes, int n_in,
                              void* d_out, int out_size, void* d_ws, size_t ws_size,
                              hipStream_t stream) {
    (void)n_in; (void)ws_size; (void)out_size;
    const float* xyz  = (const float*)d_in[0];
    const float* dvec = (const float*)d_in[1];
    const float* W[11];
    const float* B[11];
    for (int i = 0; i < 11; ++i) {
        W[i] = (const float*)d_in[2 + 2 * i];
        B[i] = (const float*)d_in[3 + 2 * i];
    }
    const int N = in_sizes[0] / 3;              // 131072
    float* out     = (float*)d_out;             // color: N*3 floats
    float* density = out + (size_t)N * 3;       // density: N floats

    // Layers routed through WMMA (L1..L10): {Kreal, Kpad, Nout}
    static const int KR[10] = { 63, 256, 256, 256, 256, 319, 256, 256, 256, 283 };
    static const int KP[10] = { 64, 256, 256, 256, 256, 320, 256, 256, 256, 288 };
    static const int NO[10] = {256, 256, 256, 256, 256, 256, 256, 256, 256, 128 };

    _Float16* Wp[10];
    size_t off = 0;                              // in halves; total ~1.19 MB
    _Float16* wsbase = (_Float16*)d_ws;
    for (int i = 0; i < 10; ++i) {
        Wp[i] = wsbase + off;
        off += (size_t)NO[i] * KP[i];
    }

    // Pack weights (stream-ordered before the main kernel; graph-capture safe).
    for (int i = 0; i < 10; ++i) {
        const int total = (NO[i] >> 4) * (KP[i] >> 5) * 256;   // dwords
        pack_weights<<<(total + 255) / 256, 256, 0, stream>>>(
            W[i], (unsigned*)Wp[i], KR[i], KP[i], NO[i]);
    }

    dim3 grid(N / ROWS), block(256);
    nerf_mlp_fused<<<grid, block, 0, stream>>>(
        xyz, dvec,
        Wp[0], Wp[1], Wp[2], Wp[3], Wp[4], Wp[5], Wp[6], Wp[7], Wp[8], Wp[9],
        B[0], B[1], B[2], B[3], B[4], B[5], B[6], B[7], B[8], B[9],
        W[8] /*W9 full fp32 for density row*/, W[10] /*W11*/, B[10] /*b11*/,
        out, density);
}
